// RelativePositionCordMessage_15135464751165
// MI455X (gfx1250) — matrix operand order repaired and checked by reading, not compiled, not run on hardware
//
#include <hip/hip_runtime.h>
#include <hip/hip_bf16.h>

#define EPS 1e-8f

typedef __attribute__((ext_vector_type(16))) __bf16          v16bf;
typedef __attribute__((ext_vector_type(8)))  float           v8f;
typedef __attribute__((ext_vector_type(16))) unsigned short  v16u;
typedef __attribute__((ext_vector_type(8)))  unsigned short  v8u;

// f32 -> bf16 round-to-nearest-even
__device__ __forceinline__ unsigned short f2bf(float f) {
    unsigned u = __builtin_bit_cast(unsigned, f);
    unsigned r = u + 0x7FFFu + ((u >> 16) & 1u);
    return (unsigned short)(r >> 16);
}

__device__ __forceinline__ float fast_silu(float v) {
    return v * __builtin_amdgcn_rcpf(1.0f + __expf(-v));
}

// 16x32 bf16 A-fragment (ISA 7.12.2) from a 64-float row (with conversion).
__device__ __forceinline__ v16bf afrag_f32(const float* row, int t, int hi) {
    const int b0 = 32 * t + 8 * hi;
    v16u u;
#pragma unroll
    for (int i = 0; i < 8; ++i) {
        u[i]     = f2bf(row[b0 + i]);
        u[8 + i] = f2bf(row[b0 + 16 + i]);
    }
    return __builtin_bit_cast(v16bf, u);
}

// Same fragment from a 64-bf16 row: two 16B loads, no VALU conversion.
__device__ __forceinline__ v16bf afrag_bf(const unsigned short* row, int t, int hi) {
    const int b0 = 32 * t + 8 * hi;
    v8u lo = *(const v8u*)(row + b0);
    v8u h8 = *(const v8u*)(row + b0 + 16);
    v16u u;
#pragma unroll
    for (int i = 0; i < 8; ++i) { u[i] = lo[i]; u[8 + i] = h8[i]; }
    return __builtin_bit_cast(v16bf, u);
}

// ---------------------------------------------------------------------------
// Pack W1 (rows 1..128), W2, W3 into bf16 B-fragments (ISA 7.12.2 32x16 B
// layout: lane holds column N = lane&15, K = 32*ktile + 8*(lane>=16) + (i&7) + 16*(i>>3)).
// ws ushort layout: [0..8191] W1 (16 tiles), [8192..12287] W2, [12288..16383] W3.
// ---------------------------------------------------------------------------
__global__ void pack_wmma_weights(const float* __restrict__ W1,
                                  const float* __restrict__ W2,
                                  const float* __restrict__ W3,
                                  unsigned short* __restrict__ out) {
    int lf = blockIdx.x * blockDim.x + threadIdx.x;
    if (lf >= 1024) return;                           // 32 tiles * 32 lanes
    int lane = lf & 31, tile = lf >> 5;
    int hi = lane >> 4, l15 = lane & 15;

    const float* W; unsigned short* dst; int kbase, n;
    if (tile < 16) {
        int t = tile >> 2; n = tile & 3; W = W1;
        kbase = (t < 2) ? (1 + 32 * t) : (65 + 32 * (t - 2));
        dst = out + tile * 512;
    } else if (tile < 24) {
        int tt = tile - 16; int t = tt >> 2; n = tt & 3;
        W = W2; kbase = 32 * t; dst = out + 8192 + tt * 512;
    } else {
        int tt = tile - 24; int t = tt >> 2; n = tt & 3;
        W = W3; kbase = 32 * t; dst = out + 12288 + tt * 512;
    }
    int col = 16 * n + l15;
#pragma unroll
    for (int i = 0; i < 16; ++i)
        dst[lane * 16 + i] = f2bf(W[(kbase + 8 * hi + (i & 7) + 16 * (i >> 3)) * 64 + col]);
}

// Pre-convert the (L2-resident) node feature table hh to bf16 (pairs packed).
__global__ void cvt_hh_bf16(const float* __restrict__ hh,
                            unsigned int* __restrict__ out, long npairs) {
    long i = (long)blockIdx.x * blockDim.x + threadIdx.x;
    if (i < npairs) {
        float a = hh[2 * i], b = hh[2 * i + 1];
        out[i] = (unsigned)f2bf(a) | ((unsigned)f2bf(b) << 16);
    }
}

// ---------------------------------------------------------------------------
// Fused EGNN edge kernel: one wave32 = 16 edges through the whole MLP.
// ---------------------------------------------------------------------------
template <bool HHBF>
__global__ __launch_bounds__(256)
void egnn_edge_kernel(const float* __restrict__ x, const float* __restrict__ hh,
                      const unsigned short* __restrict__ hhb,
                      const int* __restrict__ src_idx, const int* __restrict__ dst_idx,
                      const unsigned short* __restrict__ B1f,
                      const unsigned short* __restrict__ B2f,
                      const unsigned short* __restrict__ B3f,
                      const float* __restrict__ b1, const float* __restrict__ b2,
                      const float* __restrict__ b3,
                      const float* __restrict__ W1,   // row 0 = radial coefficients
                      const float* __restrict__ W4,
                      float* __restrict__ out_radial, float* __restrict__ out_trans,
                      float* __restrict__ out_ef, int E) {
    __shared__ __align__(16) unsigned short stageb[8][16 * 64]; // bf16 activations
    __shared__ __align__(16) float          stagef[8][16 * 64]; // f32 h3 (for W4 dot)
    __shared__ float rc[8][16][4];                              // {cdx,cdy,cdz,radial}
    __shared__ float w4s[64];

    const int wv = threadIdx.x >> 5, lane = threadIdx.x & 31;
    const int hi = lane >> 4, l15 = lane & 15;
    if (threadIdx.x < 64) w4s[threadIdx.x] = W4[threadIdx.x];

    const long e0 = (long)(blockIdx.x * 8 + wv) * 16;
    long me = e0 + l15;
    if (me >= E) me = E - 1;                  // clamp; stores are guarded

    const int si = src_idx[me], di = dst_idx[me];
    const float cx = x[si * 3 + 0] - x[di * 3 + 0];
    const float cy = x[si * 3 + 1] - x[di * 3 + 1];
    const float cz = x[si * 3 + 2] - x[di * 3 + 2];
    const float radial = cx * cx + cy * cy + cz * cz;
    if (hi == 0) {
        rc[wv][l15][0] = cx; rc[wv][l15][1] = cy;
        rc[wv][l15][2] = cz; rc[wv][l15][3] = radial;
    }

    // Layer-1 A fragments from the node-feature gathers.
    v16bf As0, As1, Ad0, Ad1;
    if (HHBF) {
        const unsigned short* hs = hhb + (long)si * 64;
        const unsigned short* hd = hhb + (long)di * 64;
        As0 = afrag_bf(hs, 0, hi); As1 = afrag_bf(hs, 1, hi);
        Ad0 = afrag_bf(hd, 0, hi); Ad1 = afrag_bf(hd, 1, hi);
    } else {
        const float* hs = hh + (long)si * 64;
        const float* hd = hh + (long)di * 64;
        As0 = afrag_f32(hs, 0, hi); As1 = afrag_f32(hs, 1, hi);
        Ad0 = afrag_f32(hd, 0, hi); Ad1 = afrag_f32(hd, 1, hi);
    }
    __syncthreads();

    const v16u* B1v = (const v16u*)B1f;
    const v16u* B2v = (const v16u*)B2f;
    const v16u* B3v = (const v16u*)B3f;

    // ---------------- Layer 1: h1 = SiLU([radial, h_src, h_dst] @ W1 + b1)
#pragma unroll
    for (int n = 0; n < 4; ++n) {
        const int col = 16 * n + l15;
        const float bias = b1[col];
        const float w1r0 = W1[col];          // W1[0][col]
        v8f acc;
#pragma unroll
        for (int g = 0; g < 8; ++g)
            acc[g] = bias + rc[wv][8 * hi + g][3] * w1r0;

        v16bf b;
        b = __builtin_bit_cast(v16bf, B1v[(0 * 4 + n) * 32 + lane]);
        acc = __builtin_amdgcn_wmma_f32_16x16x32_bf16(false, As0, false, b, (short)0, acc, false, false);
        b = __builtin_bit_cast(v16bf, B1v[(1 * 4 + n) * 32 + lane]);
        acc = __builtin_amdgcn_wmma_f32_16x16x32_bf16(false, As1, false, b, (short)0, acc, false, false);
        b = __builtin_bit_cast(v16bf, B1v[(2 * 4 + n) * 32 + lane]);
        acc = __builtin_amdgcn_wmma_f32_16x16x32_bf16(false, Ad0, false, b, (short)0, acc, false, false);
        b = __builtin_bit_cast(v16bf, B1v[(3 * 4 + n) * 32 + lane]);
        acc = __builtin_amdgcn_wmma_f32_16x16x32_bf16(false, Ad1, false, b, (short)0, acc, false, false);

#pragma unroll
        for (int g = 0; g < 8; ++g)
            stageb[wv][(8 * hi + g) * 64 + col] = f2bf(fast_silu(acc[g]));
    }
    __syncthreads();

    v16bf A20 = afrag_bf(&stageb[wv][l15 * 64], 0, hi);
    v16bf A21 = afrag_bf(&stageb[wv][l15 * 64], 1, hi);
    __syncthreads();

    // ---------------- Layer 2: edge_feature = SiLU(h1 @ W2 + b2)
#pragma unroll
    for (int n = 0; n < 4; ++n) {
        const int col = 16 * n + l15;
        const float bias = b2[col];
        v8f acc;
#pragma unroll
        for (int g = 0; g < 8; ++g) acc[g] = bias;

        v16bf b;
        b = __builtin_bit_cast(v16bf, B2v[(0 * 4 + n) * 32 + lane]);
        acc = __builtin_amdgcn_wmma_f32_16x16x32_bf16(false, A20, false, b, (short)0, acc, false, false);
        b = __builtin_bit_cast(v16bf, B2v[(1 * 4 + n) * 32 + lane]);
        acc = __builtin_amdgcn_wmma_f32_16x16x32_bf16(false, A21, false, b, (short)0, acc, false, false);

#pragma unroll
        for (int g = 0; g < 8; ++g) {
            float v = fast_silu(acc[g]);
            stageb[wv][(8 * hi + g) * 64 + col] = f2bf(v);
            const long e = e0 + 8 * hi + g;       // direct 64B-contiguous stores
            if (e < E) out_ef[e * 64 + col] = v;
        }
    }
    __syncthreads();

    v16bf A30 = afrag_bf(&stageb[wv][l15 * 64], 0, hi);
    v16bf A31 = afrag_bf(&stageb[wv][l15 * 64], 1, hi);

    // ---------------- Layer 3: h3 = SiLU(edge_feature @ W3 + b3)  (f32 stage)
#pragma unroll
    for (int n = 0; n < 4; ++n) {
        const int col = 16 * n + l15;
        const float bias = b3[col];
        v8f acc;
#pragma unroll
        for (int g = 0; g < 8; ++g) acc[g] = bias;

        v16bf b;
        b = __builtin_bit_cast(v16bf, B3v[(0 * 4 + n) * 32 + lane]);
        acc = __builtin_amdgcn_wmma_f32_16x16x32_bf16(false, A30, false, b, (short)0, acc, false, false);
        b = __builtin_bit_cast(v16bf, B3v[(1 * 4 + n) * 32 + lane]);
        acc = __builtin_amdgcn_wmma_f32_16x16x32_bf16(false, A31, false, b, (short)0, acc, false, false);

#pragma unroll
        for (int g = 0; g < 8; ++g)
            stagef[wv][(8 * hi + g) * 64 + col] = fast_silu(acc[g]);
    }
    __syncthreads();

    // ---------------- scale = tanh(h3 @ W4); radial / trans outputs.
    {
        const int r = lane >> 1, h = lane & 1;
        const float* srow = &stagef[wv][r * 64 + 32 * h];
        float s = 0.0f;
#pragma unroll
        for (int j = 0; j < 32; ++j) s += srow[j] * w4s[32 * h + j];
        s += __shfl_xor(s, 1, 32);
        const float ex = __expf(2.0f * s);                       // tanh(s)
        const float scale = 1.0f - 2.0f * __builtin_amdgcn_rcpf(ex + 1.0f);

        const long e = e0 + r;
        if (h == 0 && e < E) {
            const float rad = rc[wv][r][3];
            const float inv = __builtin_amdgcn_rcpf(sqrtf(rad) + EPS);
            out_radial[e] = rad;
            out_trans[3 * e + 0] = rc[wv][r][0] * inv * scale;
            out_trans[3 * e + 1] = rc[wv][r][1] * inv * scale;
            out_trans[3 * e + 2] = rc[wv][r][2] * inv * scale;
        }
    }
}

extern "C" void kernel_launch(void* const* d_in, const int* in_sizes, int n_in,
                              void* d_out, int out_size, void* d_ws, size_t ws_size,
                              hipStream_t stream) {
    const float* x   = (const float*)d_in[0];
    const float* hh  = (const float*)d_in[1];
    const int*   src = (const int*)d_in[2];
    const int*   dst = (const int*)d_in[3];
    const float* W1  = (const float*)d_in[4];
    const float* b1  = (const float*)d_in[5];
    const float* W2  = (const float*)d_in[6];
    const float* b2  = (const float*)d_in[7];
    const float* W3  = (const float*)d_in[8];
    const float* b3  = (const float*)d_in[9];
    const float* W4  = (const float*)d_in[10];

    const int  E     = in_sizes[2];
    const long hhlen = in_sizes[1];                   // n_nodes * 64

    unsigned short* Bf  = (unsigned short*)d_ws;      // 32 KB bf16 B-fragments
    unsigned short* hhb = (unsigned short*)((char*)d_ws + 32768);
    const size_t need = 32768 + (size_t)hhlen * 2;    // + bf16 copy of hh

    pack_wmma_weights<<<4, 256, 0, stream>>>(W1, W2, W3, Bf);

    float* out_radial = (float*)d_out;                // [E]
    float* out_trans  = out_radial + E;               // [E,3]
    float* out_ef     = out_radial + 4 * (long)E;     // [E,64]

    const int blocks = (E + 127) / 128;               // 8 waves * 16 edges / block

    if (ws_size >= need) {
        const long npairs = hhlen / 2;
        cvt_hh_bf16<<<(int)((npairs + 255) / 256), 256, 0, stream>>>(
            hh, (unsigned int*)hhb, npairs);
        egnn_edge_kernel<true><<<blocks, 256, 0, stream>>>(
            x, hh, hhb, src, dst, Bf, Bf + 8192, Bf + 12288,
            b1, b2, b3, W1, W4, out_radial, out_trans, out_ef, E);
    } else {
        egnn_edge_kernel<false><<<blocks, 256, 0, stream>>>(
            x, hh, hhb, src, dst, Bf, Bf + 8192, Bf + 12288,
            b1, b2, b3, W1, W4, out_radial, out_trans, out_ef, E);
    }
}
